// CausalSelectiveSelfAttentionForInference_19859928776996
// MI455X (gfx1250) — matrix operand order repaired and checked by reading, not compiled, not run on hardware
//
#include <hip/hip_runtime.h>
#include <stdint.h>

#define B_N 2
#define T_N 2048
#define C_N 768
#define H_N 12
#define D_N 64
#define C3_N 2304
#define KEEPW (T_N / 32)   // 64 words of keep-bits per row
#define EFFK_CAP 408       // memory_budget(409) - 1

typedef __attribute__((ext_vector_type(16))) __bf16 bf16x16;
typedef __attribute__((ext_vector_type(8)))  float  f32x8;

union BF16Frag { bf16x16 v; uint4 u[2]; };

__device__ __forceinline__ f32x8 wmma_bf16(bf16x16 a, bf16x16 b, f32x8 c) {
    // 8 args: (neg_a, A, neg_b, B, c_mod, C, reuse_a, reuse_b)
    return __builtin_amdgcn_wmma_f32_16x16x32_bf16(false, a, false, b, (short)0, c, false, false);
}

// A-fragment (16x32 bf16): lanes 0-15 row M=lane K{0..7,16..23}; lanes 16-31 same row, K{8..15,24..31}
__device__ __forceinline__ bf16x16 load_a_frag(const __bf16* base, int ld, int row, int k0, int hf) {
    const __bf16* p = base + (size_t)row * ld + k0;
    BF16Frag f;
    f.u[0] = *reinterpret_cast<const uint4*>(p + hf * 8);
    f.u[1] = *reinterpret_cast<const uint4*>(p + 16 + hf * 8);
    return f.v;
}

// B-fragment (32x16 bf16) from column-contiguous ("transposed") storage: element (k,n) at baseT[n*ld + k]
// lanes 0-15: col N=lane, K{0..7,16..23}; lanes 16-31: col N=lane-16, K{8..15,24..31}
__device__ __forceinline__ bf16x16 load_b_frag(const __bf16* baseT, int ld, int col, int k0, int hf) {
    const __bf16* p = baseT + (size_t)col * ld + k0;
    BF16Frag f;
    f.u[0] = *reinterpret_cast<const uint4*>(p + hf * 8);
    f.u[1] = *reinterpret_cast<const uint4*>(p + 16 + hf * 8);
    return f.v;
}

// ---------------------------------------------------------------- pack kernels
__global__ void cvt_f32_bf16_kernel(const float* __restrict__ src, __bf16* __restrict__ dst, int n) {
    int idx = blockIdx.x * blockDim.x + threadIdx.x;
    if (idx < n) dst[idx] = (__bf16)src[idx];
}

__global__ void transpose_to_bf16_kernel(const float* __restrict__ w, __bf16* __restrict__ wT, int K, int N) {
    int idx = blockIdx.x * blockDim.x + threadIdx.x;
    if (idx < K * N) {
        int kk = idx / N, n = idx - kk * N;
        wT[(size_t)n * K + kk] = (__bf16)w[idx];
    }
}

// ---------------------------------------------------------------- QKV projection (WMMA)
// out[m, n] = x[m, :] @ w_attn[:, n] + b_attn[n]; scatter into q (scaled), k, vT
__global__ void qkv_gemm_kernel(const __bf16* __restrict__ xbf, const __bf16* __restrict__ wT,
                                const float* __restrict__ b_attn,
                                __bf16* __restrict__ q, __bf16* __restrict__ k, __bf16* __restrict__ vT) {
    int lane = threadIdx.x, ln = lane & 15, hf = lane >> 4;
    int tn = blockIdx.x, tm = blockIdx.y;
    int row0 = tm * 16;
    f32x8 c[4] = {};
    for (int k0 = 0; k0 < C_N; k0 += 32) {
        bf16x16 a = load_a_frag(xbf, C_N, row0 + ln, k0, hf);
#pragma unroll
        for (int t = 0; t < 4; ++t) {
            bf16x16 bfr = load_b_frag(wT, C_N, tn * 64 + t * 16 + ln, k0, hf);
            c[t] = wmma_bf16(a, bfr, c[t]);
        }
    }
#pragma unroll
    for (int t = 0; t < 4; ++t) {
        int gcol = tn * 64 + t * 16 + ln;
        float bias = b_attn[gcol];
        int sec = gcol / C_N;              // 0=q 1=k 2=v
        int inner = gcol - sec * C_N;
        int h = inner >> 6, d = inner & 63;
#pragma unroll
        for (int r = 0; r < 8; ++r) {
            int grow = row0 + r + 8 * hf;
            int b = grow >> 11, tt = grow & (T_N - 1);
            float v = c[t][r] + bias;
            if (sec == 0) { v *= 0.125f; q[((size_t)(b * H_N + h) * T_N + tt) * D_N + d] = (__bf16)v; }
            else if (sec == 1) { k[((size_t)(b * H_N + h) * T_N + tt) * D_N + d] = (__bf16)v; }
            else { vT[((size_t)(b * H_N + h) * D_N + d) * T_N + tt] = (__bf16)v; }
        }
    }
}

// ---------------------------------------------------------------- FF: column prefix sum of masked relu(head0 scores)
// FF[b, i, j] = sum_{i' < i} S[b, i', j],  S = relu(q0.k0) with (j<=i', j!=0, j!=i')
__global__ void ff_kernel(const __bf16* __restrict__ q, const __bf16* __restrict__ k,
                          float* __restrict__ FF) {
    __shared__ float satt[256];
    int lane = threadIdx.x, ln = lane & 15, hf = lane >> 4;
    int jt = blockIdx.x, b = blockIdx.y;
    const __bf16* qh = q + (size_t)(b * H_N) * T_N * D_N;   // head 0
    const __bf16* kh = k + (size_t)(b * H_N) * T_N * D_N;
    int j = jt * 16 + lane;    // lanes 0..15 each own one key column
    float colsum = 0.f;
    for (int it = 0; it < T_N / 16; ++it) {
        f32x8 c = {};
        bf16x16 a0 = load_a_frag(qh, D_N, it * 16 + ln, 0, hf);
        bf16x16 b0 = load_b_frag(kh, D_N, jt * 16 + ln, 0, hf);
        c = wmma_bf16(a0, b0, c);
        bf16x16 a1 = load_a_frag(qh, D_N, it * 16 + ln, 32, hf);
        bf16x16 b1 = load_b_frag(kh, D_N, jt * 16 + ln, 32, hf);
        c = wmma_bf16(a1, b1, c);
#pragma unroll
        for (int r = 0; r < 8; ++r) satt[(r + 8 * hf) * 16 + ln] = c[r];
        __syncthreads();
        if (lane < 16) {
            float* ffp = FF + ((size_t)b * T_N + it * 16) * T_N + j;
#pragma unroll
            for (int r = 0; r < 16; ++r) {
                int i = it * 16 + r;
                ffp[(size_t)r * T_N] = colsum;   // FF row i gets sum through row i-1 (shift)
                float s = satt[r * 16 + lane];
                if (j <= i && j != 0 && j != i && s > 0.f) colsum += s;
            }
        }
        __syncthreads();
    }
}

// ---------------------------------------------------------------- stable top-eff_k selection (radix select on float bits)
__global__ void select_kernel(const float* __restrict__ FF, unsigned* __restrict__ keep) {
    __shared__ unsigned hist[16];
    __shared__ unsigned scnt;
    __shared__ unsigned wsum[8];
    __shared__ unsigned runcnt;
    int i = blockIdx.x, b = blockIdx.y;
    int tid = threadIdx.x;
    const float* row = FF + ((size_t)b * T_N + i) * T_N;
    unsigned* krow = keep + ((size_t)b * T_N + i) * KEEPW;
    int L = i + 1;
    int effk = L < EFFK_CAP ? L : EFFK_CAP;
    if (L <= effk) {                       // keep all causal positions
        for (int w = tid; w < KEEPW; w += (int)blockDim.x) {
            int lo = w * 32;
            unsigned m;
            if (i >= lo + 31) m = 0xFFFFFFFFu;
            else if (i < lo) m = 0u;
            else m = (1u << (i - lo + 1)) - 1u;
            krow[w] = m;
        }
        return;
    }
    // FF >= 0 -> uint bit pattern is order-preserving
    unsigned prefix = 0, pmask = 0;
    int kk = effk;
    for (int pass = 7; pass >= 0; --pass) {
        int shift = pass * 4;
        if (tid < 16) hist[tid] = 0u;
        __syncthreads();
        for (int j = tid; j < L; j += (int)blockDim.x) {
            unsigned u = __float_as_uint(row[j]);
            if ((u & pmask) == prefix) atomicAdd(&hist[(u >> shift) & 15], 1u);
        }
        __syncthreads();
        unsigned cum = 0; int sel = 15;
        for (int d = 0; d < 16; ++d) {
            unsigned h = hist[d];
            if (cum + h >= (unsigned)kk) { sel = d; break; }
            cum += h;
        }
        kk -= (int)cum;
        prefix |= ((unsigned)sel) << shift;
        pmask |= 0xFu << shift;
        __syncthreads();
    }
    if (tid == 0) { scnt = 0u; runcnt = 0u; }
    __syncthreads();
    unsigned local = 0;
    for (int j = tid; j < L; j += (int)blockDim.x)
        if (__float_as_uint(row[j]) < prefix) local++;
    atomicAdd(&scnt, local);
    __syncthreads();
    int slots = effk - (int)scnt;          // >= 1 slots for ties at threshold value
    int lanei = tid & 31, wid = tid >> 5;
    for (int base = 0; base < T_N; base += (int)blockDim.x) {
        int j = base + tid;
        bool inr = (j < L);
        unsigned u = inr ? __float_as_uint(row[j]) : 0xFFFFFFFFu;
        bool eq = inr && (u == prefix);
        bool lt = inr && (u < prefix);
        unsigned long long bal = __ballot(eq);
        unsigned myex = (unsigned)__popcll(bal & ((1ull << lanei) - 1ull));
        if (lanei == 0) wsum[wid] = (unsigned)__popcll(bal);
        __syncthreads();
        unsigned woff = 0;
        for (int w = 0; w < wid; ++w) woff += wsum[w];
        unsigned tp = runcnt + woff + myex;       // stable tie rank (matches stable argsort)
        bool kp = lt || (eq && (tp < (unsigned)slots));
        unsigned long long kb = __ballot(kp);
        if (lanei == 0) krow[(base >> 5) + wid] = (unsigned)kb;
        __syncthreads();
        if (tid == 0) {
            unsigned tot = 0;
            for (int w = 0; w < 8; ++w) tot += wsum[w];
            runcnt += tot;
        }
        __syncthreads();
    }
    if (tid == 0) krow[i >> 5] |= (1u << (i & 31));   // always keep self
}

// ---------------------------------------------------------------- flash attention with FF decay + keep mask (WMMA)
__global__ void attn_kernel(const __bf16* __restrict__ q, const __bf16* __restrict__ k,
                            const __bf16* __restrict__ vT, const float* __restrict__ FF,
                            const unsigned* __restrict__ keep, __bf16* __restrict__ ybf) {
    __shared__ float ldsP[16 * 32];
    int lane = threadIdx.x, ln = lane & 15, hf = lane >> 4;
    int it = blockIdx.x, h = blockIdx.y, b = blockIdx.z;
    int i0 = it * 16;
    const __bf16* qh = q + (size_t)(b * H_N + h) * T_N * D_N;
    const __bf16* kh = k + (size_t)(b * H_N + h) * T_N * D_N;
    const __bf16* vh = vT + (size_t)(b * H_N + h) * D_N * T_N;
    const float* ffb = FF + (size_t)b * T_N * T_N;
    const unsigned* kpb = keep + (size_t)b * T_N * KEEPW;

    bf16x16 aq0 = load_a_frag(qh, D_N, i0 + ln, 0, hf);
    bf16x16 aq1 = load_a_frag(qh, D_N, i0 + ln, 32, hf);

    float mrow[8], lrow[8];
    f32x8 acc[4] = {};
#pragma unroll
    for (int r = 0; r < 8; ++r) { mrow[r] = -1e30f; lrow[r] = 0.f; }

    for (int jc = 0; jc < i0 + 16; jc += 32) {
        f32x8 s0 = {}, s1 = {};
        {
            bf16x16 bk = load_b_frag(kh, D_N, jc + ln, 0, hf);
            s0 = wmma_bf16(aq0, bk, s0);
            bk = load_b_frag(kh, D_N, jc + ln, 32, hf);
            s0 = wmma_bf16(aq1, bk, s0);
            bk = load_b_frag(kh, D_N, jc + 16 + ln, 0, hf);
            s1 = wmma_bf16(aq0, bk, s1);
            bk = load_b_frag(kh, D_N, jc + 16 + ln, 32, hf);
            s1 = wmma_bf16(aq1, bk, s1);
        }
        float att0[8], att1[8];
#pragma unroll
        for (int r = 0; r < 8; ++r) {
            int i = i0 + r + 8 * hf;
            int j0c = jc + ln;
            int j1c = jc + 16 + ln;
            float a0 = s0[r] - ffb[(size_t)i * T_N + j0c];
            float a1 = s1[r] - ffb[(size_t)i * T_N + j1c];
            unsigned kw0 = kpb[(size_t)i * KEEPW + (j0c >> 5)];
            unsigned kw1 = kpb[(size_t)i * KEEPW + (j1c >> 5)];
            bool v0 = (j0c <= i) && ((kw0 >> (j0c & 31)) & 1u);
            bool v1 = (j1c <= i) && ((kw1 >> (j1c & 31)) & 1u);
            att0[r] = v0 ? a0 : -1e30f;
            att1[r] = v1 ? a1 : -1e30f;
        }
#pragma unroll
        for (int r = 0; r < 8; ++r) {
            float rm = fmaxf(att0[r], att1[r]);
            rm = fmaxf(rm, __shfl_xor(rm, 1, 32));
            rm = fmaxf(rm, __shfl_xor(rm, 2, 32));
            rm = fmaxf(rm, __shfl_xor(rm, 4, 32));
            rm = fmaxf(rm, __shfl_xor(rm, 8, 32));
            float nm = fmaxf(mrow[r], rm);
            float e  = mrow[r] - nm;
            float sc = (e < -80.f) ? 0.f : __expf(e);
            float e0 = att0[r] - nm;
            float e1 = att1[r] - nm;
            float p0 = (e0 < -80.f) ? 0.f : __expf(e0);
            float p1 = (e1 < -80.f) ? 0.f : __expf(e1);
            float rs = p0 + p1;
            rs += __shfl_xor(rs, 1, 32);
            rs += __shfl_xor(rs, 2, 32);
            rs += __shfl_xor(rs, 4, 32);
            rs += __shfl_xor(rs, 8, 32);
            lrow[r] = lrow[r] * sc + rs;
            mrow[r] = nm;
#pragma unroll
            for (int dt = 0; dt < 4; ++dt) acc[dt][r] = acc[dt][r] * sc;
            ldsP[(r + 8 * hf) * 32 + ln] = p0;
            ldsP[(r + 8 * hf) * 32 + 16 + ln] = p1;
        }
        __syncthreads();
        bf16x16 pa;
        {
            const float* pr = &ldsP[ln * 32];
#pragma unroll
            for (int e2 = 0; e2 < 8; ++e2) pa[e2] = (__bf16)pr[hf * 8 + e2];
#pragma unroll
            for (int e2 = 0; e2 < 8; ++e2) pa[8 + e2] = (__bf16)pr[16 + hf * 8 + e2];
        }
        __syncthreads();
#pragma unroll
        for (int dt = 0; dt < 4; ++dt) {
            bf16x16 bv = load_b_frag(vh, T_N, dt * 16 + ln, jc, hf);
            acc[dt] = wmma_bf16(pa, bv, acc[dt]);
        }
    }
#pragma unroll
    for (int r = 0; r < 8; ++r) {
        int i = i0 + r + 8 * hf;
        float inv = 1.f / lrow[r];
#pragma unroll
        for (int dt = 0; dt < 4; ++dt) {
            ybf[((size_t)b * T_N + i) * C_N + h * D_N + dt * 16 + ln] = (__bf16)(acc[dt][r] * inv);
        }
    }
}

// ---------------------------------------------------------------- output projection (WMMA), fp32 out
__global__ void proj_gemm_kernel(const __bf16* __restrict__ ybf, const __bf16* __restrict__ wT,
                                 const float* __restrict__ b_proj, float* __restrict__ out) {
    int lane = threadIdx.x, ln = lane & 15, hf = lane >> 4;
    int tn = blockIdx.x, tm = blockIdx.y;
    int row0 = tm * 16;
    f32x8 c[4] = {};
    for (int k0 = 0; k0 < C_N; k0 += 32) {
        bf16x16 a = load_a_frag(ybf, C_N, row0 + ln, k0, hf);
#pragma unroll
        for (int t = 0; t < 4; ++t) {
            bf16x16 bfr = load_b_frag(wT, C_N, tn * 64 + t * 16 + ln, k0, hf);
            c[t] = wmma_bf16(a, bfr, c[t]);
        }
    }
#pragma unroll
    for (int t = 0; t < 4; ++t) {
        int col = tn * 64 + t * 16 + ln;
        float bias = b_proj[col];
#pragma unroll
        for (int r = 0; r < 8; ++r) {
            int grow = row0 + r + 8 * hf;
            out[(size_t)grow * C_N + col] = c[t][r] + bias;
        }
    }
}

// ---------------------------------------------------------------- launch
extern "C" void kernel_launch(void* const* d_in, const int* in_sizes, int n_in,
                              void* d_out, int out_size, void* d_ws, size_t ws_size,
                              hipStream_t stream) {
    (void)in_sizes; (void)n_in; (void)out_size; (void)ws_size;
    const float* x      = (const float*)d_in[0];
    const float* w_attn = (const float*)d_in[1];
    const float* b_attn = (const float*)d_in[2];
    const float* w_proj = (const float*)d_in[3];
    const float* b_proj = (const float*)d_in[4];
    float* out = (float*)d_out;
    char* ws = (char*)d_ws;

    constexpr size_t SZ_XBF = (size_t)B_N * T_N * C_N * 2;
    constexpr size_t SZ_WAT = (size_t)C_N * C3_N * 2;
    constexpr size_t SZ_WPT = (size_t)C_N * C_N * 2;
    constexpr size_t SZ_QKV = (size_t)B_N * H_N * T_N * D_N * 2;
    constexpr size_t SZ_YBF = (size_t)B_N * T_N * C_N * 2;
    constexpr size_t SZ_FF  = (size_t)B_N * T_N * T_N * 4;

    size_t o = 0;
    __bf16* xbf    = (__bf16*)(ws + o); o += SZ_XBF;
    __bf16* wattnT = (__bf16*)(ws + o); o += SZ_WAT;
    __bf16* wprojT = (__bf16*)(ws + o); o += SZ_WPT;
    __bf16* qb     = (__bf16*)(ws + o); o += SZ_QKV;
    __bf16* kb     = (__bf16*)(ws + o); o += SZ_QKV;
    __bf16* vTb    = (__bf16*)(ws + o); o += SZ_QKV;
    __bf16* ybf    = (__bf16*)(ws + o); o += SZ_YBF;
    float*    FF   = (float*)(ws + o);  o += SZ_FF;
    unsigned* keep = (unsigned*)(ws + o);

    int nx = B_N * T_N * C_N;
    cvt_f32_bf16_kernel<<<(nx + 255) / 256, 256, 0, stream>>>(x, xbf, nx);
    transpose_to_bf16_kernel<<<(C_N * C3_N + 255) / 256, 256, 0, stream>>>(w_attn, wattnT, C_N, C3_N);
    transpose_to_bf16_kernel<<<(C_N * C_N + 255) / 256, 256, 0, stream>>>(w_proj, wprojT, C_N, C_N);
    qkv_gemm_kernel<<<dim3(C3_N / 64, (B_N * T_N) / 16), 32, 0, stream>>>(xbf, wattnT, b_attn, qb, kb, vTb);
    ff_kernel<<<dim3(T_N / 16, B_N), 32, 0, stream>>>(qb, kb, FF);
    select_kernel<<<dim3(T_N, B_N), 256, 0, stream>>>(FF, keep);
    attn_kernel<<<dim3(T_N / 16, H_N, B_N), 32, 0, stream>>>(qb, kb, vTb, FF, keep, ybf);
    proj_gemm_kernel<<<dim3(C_N / 64, (B_N * T_N) / 16), 32, 0, stream>>>(ybf, wprojT, b_proj, out);
}
